// SphericalHarmonicsLayer_17806934409297
// MI455X (gfx1250) — compile-verified, matrix operand order checked by reading, not run on hardware
//
#include <hip/hip_runtime.h>

typedef _Float16 v16h __attribute__((ext_vector_type(16)));
typedef float    v8f  __attribute__((ext_vector_type(8)));

#define H_DIM 256
#define HT    16      // H_DIM / 16 h-tiles

// One wave = one 16(row=b*N+n) x 16(h) tile.
// Linear(1,H) layers done on the matrix pipe: v_wmma_f32_16x16x32_f16 with
// only the K=0 slice populated (A = x scalar, B = weight row, C = bias).
__global__ __launch_bounds__(256)
void sh_l3_kernel(const float* __restrict__ x,
                  const float* __restrict__ w_th, const float* __restrict__ b_th,
                  const float* __restrict__ w_ph, const float* __restrict__ b_ph,
                  const float* __restrict__ w_tm, const float* __restrict__ b_tm,
                  float* __restrict__ out)
{
    const int gid  = blockIdx.x * 256 + threadIdx.x;
    const int wave = gid >> 5;
    const int lane = threadIdx.x & 31;
    const int l15  = lane & 15;
    const bool lo  = lane < 16;

    const int ht    = wave & (HT - 1);
    const int rt    = wave >> 4;          // tile over rows r = b*N + n
    const int hbase = ht << 4;
    const int rbase = rt << 4;
    const int h     = hbase + l15;        // D-matrix column owned by this lane

    // Per-column parameters (lanes 16-31 mirror 0-15; broadcast loads hit cache)
    const float wth = w_th[h], bth = b_th[h];
    const float wph = w_ph[h], bph = b_ph[h];
    const float wtm = w_tm[h], btm = b_tm[h];

    // Per-row inputs: lane l15 supplies row rbase+l15 into the A matrix K=0 slot
    const float* xp = x + (size_t)(rbase + l15) * 3;
    const float x0 = xp[0], x1 = xp[1], x2 = xp[2];

    const _Float16 z = (_Float16)0.0f;
    v16h a_th = {}, a_ph = {}, a_tm = {};
    v16h wb_th = {}, wb_ph = {}, wb_tm = {};
    // 16-bit A 16x32 layout: lanes 0-15 hold K=0..1 in VGPR0 (low half = K=0).
    a_th[0]  = lo ? (_Float16)x0  : z;
    a_ph[0]  = lo ? (_Float16)x1  : z;
    a_tm[0]  = lo ? (_Float16)x2  : z;
    // B 32x16: K=0 row (N = lane) in low half of first VGPR, lanes 0-15.
    wb_th[0] = lo ? (_Float16)wth : z;
    wb_ph[0] = lo ? (_Float16)wph : z;
    wb_tm[0] = lo ? (_Float16)wtm : z;

    v8f c_th, c_ph, c_tm;
#pragma unroll
    for (int i = 0; i < 8; ++i) { c_th[i] = bth; c_ph[i] = bph; c_tm[i] = btm; }

    // D = x*w + bias on the matrix pipe (EXEC all-1s: no divergence, exact grid)
    v8f th = __builtin_amdgcn_wmma_f32_16x16x32_f16(false, a_th, false, wb_th,
                                                    (short)0, c_th, false, false);
    v8f ph = __builtin_amdgcn_wmma_f32_16x16x32_f16(false, a_ph, false, wb_ph,
                                                    (short)0, c_ph, false, false);
    v8f tm = __builtin_amdgcn_wmma_f32_16x16x32_f16(false, a_tm, false, wb_tm,
                                                    (short)0, c_tm, false, false);

    const int rofs = lo ? 0 : 8;          // lanes 16-31 hold rows M=8..15

#pragma unroll
    for (int i = 0; i < 8; ++i) {
        const float theta = th[i];
        const float phi   = ph[i];
        const float tval  = tm[i];

        // Hardware transcendentals (v_cos_f32 / v_sin_f32 / v_sqrt_f32)
        const float ct  = __cosf(theta);
        float s2 = 1.0f - ct * ct; s2 = s2 < 0.0f ? 0.0f : s2;
        const float s   = __builtin_amdgcn_sqrtf(s2);
        const float sp  = __sinf(phi);
        const float cp  = __cosf(phi);
        const float s2p = 2.0f * sp * cp;
        const float c2p = 2.0f * cp * cp - 1.0f;
        const float s3p = sp * (3.0f - 4.0f * sp * sp);
        const float c3p = cp * (4.0f * cp * cp - 3.0f);
        const float ct2 = ct * ct;

        float o[17];
        o[0]  = 0.28209479177387814f;                       // Y00
        o[1]  = -0.4886025119029199f * s * sp;              // Y1,-1 (P11=-s)
        o[2]  =  0.4886025119029199f * ct;                  // Y1,0
        o[3]  = -0.4886025119029199f * s * cp;              // Y1,1
        o[4]  =  0.5462742152960396f * s2 * s2p;            // Y2,-2 (3s^2)
        o[5]  = -1.0925484305920792f * ct * s * sp;         // Y2,-1 (-3xs)
        o[6]  =  0.9461746957575601f * ct2 - 0.31539156525252005f; // Y2,0
        o[7]  = -1.0925484305920792f * ct * s * cp;         // Y2,1
        o[8]  =  0.5462742152960396f * s2 * c2p;            // Y2,2
        o[9]  = -0.5900435899266435f * s * s2 * s3p;        // Y3,-3 (-15s^3)
        o[10] =  1.4453057213202770f * ct * s2 * s2p;       // Y3,-2 (15xs^2)
        o[11] = -0.4570457994644658f * (5.0f * ct2 - 1.0f) * s * sp; // Y3,-1
        o[12] =  ct * (1.865881662950577f * ct2 - 1.1195289977703462f); // Y3,0
        o[13] = -0.4570457994644658f * (5.0f * ct2 - 1.0f) * s * cp; // Y3,1
        o[14] =  1.4453057213202770f * ct * s2 * c2p;       // Y3,2
        o[15] = -0.5900435899266435f * s * s2 * c3p;        // Y3,3
        o[16] =  tval;                                      // time

        const int r = rbase + rofs + i;                     // row = b*N + n
        float* op = out + ((size_t)r * H_DIM + h) * 17;
        // 285 MB write-once stream > 192 MB L2: non-temporal stores.
#pragma unroll
        for (int k = 0; k < 17; ++k)
            __builtin_nontemporal_store(o[k], op + k);
    }
}

extern "C" void kernel_launch(void* const* d_in, const int* in_sizes, int n_in,
                              void* d_out, int out_size, void* d_ws, size_t ws_size,
                              hipStream_t stream) {
    (void)n_in; (void)d_ws; (void)ws_size; (void)out_size;
    const float* x    = (const float*)d_in[0];
    const float* wth  = (const float*)d_in[1];
    const float* bth  = (const float*)d_in[2];
    const float* wph  = (const float*)d_in[3];
    const float* bph  = (const float*)d_in[4];
    const float* wtm  = (const float*)d_in[5];
    const float* btm  = (const float*)d_in[6];
    float* out = (float*)d_out;

    const int R     = in_sizes[0] / 3;          // B*N = 16384 rows
    const int waves = (R >> 4) * HT;            // 16x16 tiles -> 16384 waves
    const int threads = waves * 32;             // 524288
    sh_l3_kernel<<<threads / 256, 256, 0, stream>>>(x, wth, bth, wph, bph, wtm, btm, out);
}
